// GGCN_raf_42838003810872
// MI455X (gfx1250) — matrix operand order repaired and checked by reading, not compiled, not run on hardware
//
#include <hip/hip_runtime.h>
#include <hip/hip_bf16.h>
#include <cstdint>
#include <math.h>

// ---------------------------------------------------------------------------
// GGCN forward for MI455X (gfx1250, wave32, WMMA f16 16x16x32)
// N=4096, NFEAT=512, NHID=64, NCLS=16
// ---------------------------------------------------------------------------

typedef _Float16 v16h __attribute__((ext_vector_type(16)));
typedef _Float16 v8h  __attribute__((ext_vector_type(8)));
typedef float    v8f  __attribute__((ext_vector_type(8)));

#define NROWS 4096
#define EPSF  1e-9f
#define RFLOOR 3.16228e-5f   // sqrt(EPSF): product of two floors ~= EPSF

__device__ __forceinline__ v8f wmma_f16f32(v16h a, v16h b, v8f c) {
#if defined(__gfx1250__)
    return __builtin_amdgcn_wmma_f32_16x16x32_f16(
        /*neg_a=*/false, a, /*neg_b=*/false, b,
        /*c_mod=*/(short)0, c, /*reuse_a=*/false, /*reuse_b=*/false);
#else
    (void)a; (void)b;
    return c;
#endif
}

__device__ __forceinline__ float fast_rcp(float x) {
#if defined(__AMDGCN__)
    return __builtin_amdgcn_rcpf(x);
#else
    return 1.0f / x;
#endif
}

// A fragment (16x32 f16): lane m = lane&15; lanes 0-15 K=kb+0..7 (V0-3) and
// kb+16..23 (V4-7); lanes 16-31 K=kb+8..15 and kb+24..31.
__device__ __forceinline__ v16h load_frag_a(const _Float16* p, int hs) {
    const _Float16* q = p + hs * 8;
    v8h lo = *(const v8h*)q;
    v8h hi = *(const v8h*)(q + 16);
    v16h a;
#pragma unroll
    for (int i = 0; i < 8; ++i) { a[i] = lo[i]; a[i + 8] = hi[i]; }
    return a;
}

// B fragment (32x16 f16): lane n = lane&15; lanes 0-15 K=kb+0..7 and
// kb+16..23; lanes 16-31 K=kb+8..15 and kb+24..31.
__device__ __forceinline__ v16h load_frag_b(const _Float16* p, int hs) {
    const _Float16* q = p + hs * 8;
    v8h lo = *(const v8h*)q;
    v8h hi = *(const v8h*)(q + 16);
    v16h b;
#pragma unroll
    for (int i = 0; i < 8; ++i) { b[i] = lo[i]; b[i + 8] = hi[i]; }
    return b;
}

// fast softplus: max(z,0) + ln(1 + e^-|z|), native exp/log
__device__ __forceinline__ float softplus_fast(float z) {
    return fmaxf(z, 0.f) + __logf(1.f + __expf(-fabsf(z)));
}
__device__ __forceinline__ float elu_f(float z) {
    return z > 0.f ? z : expm1f(z);
}

// ---------------------------------------------------------------------------
// Small tiled f32 GEMM: out[r][c] = act(dot(H[r,:NIN], W[c,:NIN]) + b[c])
// 64 rows per block, 256 threads. ACT: 0=none, 1=elu.
// ---------------------------------------------------------------------------
template <int NIN, int NF, int ACT>
__global__ void __launch_bounds__(256) gemm_bias_kernel(
    const float* __restrict__ H, const float* __restrict__ W,
    const float* __restrict__ b, float* __restrict__ out) {
    constexpr int KC  = 64;
    constexpr int CPT = NF / 4;
    __shared__ float sH[64][KC + 1];
    __shared__ float sW[NF][KC + 1];
    const int tid   = threadIdx.x;
    const int r0    = tid & 63;
    const int cg    = tid >> 6;
    const int rbase = blockIdx.x * 64;
    float acc[CPT];
#pragma unroll
    for (int c = 0; c < CPT; ++c) acc[c] = 0.f;
    for (int k0 = 0; k0 < NIN; k0 += KC) {
        for (int idx = tid; idx < 64 * KC; idx += 256) {
            int rr = idx >> 6, kk = idx & 63;
            sH[rr][kk] = H[(size_t)(rbase + rr) * NIN + k0 + kk];
        }
        for (int idx = tid; idx < NF * KC; idx += 256) {
            int rr = idx >> 6, kk = idx & 63;
            sW[rr][kk] = W[(size_t)rr * NIN + k0 + kk];
        }
        __syncthreads();
#pragma unroll 8
        for (int kk = 0; kk < KC; ++kk) {
            float hv = sH[r0][kk];
#pragma unroll
            for (int c = 0; c < CPT; ++c) acc[c] += hv * sW[cg * CPT + c][kk];
        }
        __syncthreads();
    }
#pragma unroll
    for (int c = 0; c < CPT; ++c) {
        int   col = cg * CPT + c;
        float v   = acc[c] + b[col];
        if (ACT == 1) v = elu_f(v);
        out[(size_t)(rbase + r0) * NF + col] = v;
    }
}

// ---------------------------------------------------------------------------
// Pack Wh to f16 (normal layout, zero-padded to WHLD cols) AND transposed
// layout whT16[feat][row] (ld=NROWS), plus invroot = 1/max(||Wh_r||, floor).
// ---------------------------------------------------------------------------
template <int NF, int WHLD>
__global__ void __launch_bounds__(256) postwh_kernel(
    const float* __restrict__ whf32, _Float16* __restrict__ whf16,
    _Float16* __restrict__ whT16, float* __restrict__ invroot) {
    int r = blockIdx.x * blockDim.x + threadIdx.x;
    if (r >= NROWS) return;
    float s = 0.f;
#pragma unroll
    for (int c = 0; c < NF; ++c) {
        float v = whf32[(size_t)r * NF + c];
        s += v * v;
        _Float16 h = (_Float16)v;
        whf16[(size_t)r * WHLD + c] = h;
        whT16[(size_t)c * NROWS + r] = h;   // coalesced across threads
    }
#pragma unroll
    for (int c = NF; c < WHLD; ++c) whf16[(size_t)r * WHLD + c] = (_Float16)0.f;
    invroot[r] = fast_rcp(fmaxf(sqrtf(s), RFLOOR));
}

__global__ void __launch_bounds__(256) diag_kernel(
    const float* __restrict__ adj, float* __restrict__ dg) {
    int i = blockIdx.x * blockDim.x + threadIdx.x;
    if (i < NROWS) dg[i] = adj[(size_t)i * NROWS + i];
}

__global__ void __launch_bounds__(256) logsoftmax_kernel(float* __restrict__ o) {
    int r = blockIdx.x * blockDim.x + threadIdx.x;
    if (r >= NROWS) return;
    float* p = o + (size_t)r * 16;
    float  m = p[0];
#pragma unroll
    for (int c = 1; c < 16; ++c) m = fmaxf(m, p[c]);
    float s = 0.f;
#pragma unroll
    for (int c = 0; c < 16; ++c) s += __expf(p[c] - m);
    float l = m + logf(s);
#pragma unroll
    for (int c = 0; c < 16; ++c) p[c] = p[c] - l;
}

// ---------------------------------------------------------------------------
// Fused GGCN layer (flash-attention style).
//   E   = Wh_i @ Wh_j^T                         (WMMA, K = NF padded)
//   att = (E * invI * invJ) * adj * softplus(dc0*deg + dc1), diag 0
//   C   = c0*relu(att) + c1*min(att,0)  ->  f16 tile in LDS
//   acc += C @ Wh_j                             (WMMA, K = j-tile)
//   out = s * (acc + c2*Wh_i)
// MODE 0: prev[i] = coeff*elu(out) + prev[i];  MODE 1: outraw[i] = out.
// Block: 256 thr (8 waves), TI=32 output rows, sweep j in TJ=64 tiles.
// ---------------------------------------------------------------------------
template <int NF, int MODE>
__global__ void __launch_bounds__(256) ggcn_layer_kernel(
    const float* __restrict__ adj, const float* __restrict__ adjdiag,
    const float* __restrict__ whf32, const _Float16* __restrict__ whf16,
    const _Float16* __restrict__ whT16, const float* __restrict__ invroot,
    const float* __restrict__ dcP, const float* __restrict__ cfP,
    const float* __restrict__ spP,
    float* __restrict__ prev, float* __restrict__ outraw, float coeff) {
    constexpr int TI   = 32;
    constexpr int TJ   = 64;
    constexpr int KP   = (NF == 64) ? 64 : 32;  // E GEMM K (padded features)
    constexpr int WHLD = KP;                    // leading dim of whf16
    constexpr int SWL  = KP + 8;                // LDS stride (16B aligned rows)
    constexpr int SCL  = TJ + 8;                // LDS stride for C / WhT tiles
    constexpr int FT   = NF / 16;               // feature tiles (4 or 1)
    constexpr int KCH  = KP / 32;               // E k-chunks (2 or 1)
    constexpr int NJ   = NROWS / TJ;

    __shared__ __align__(16) _Float16 sWhi[TI][SWL];   // i-rows x feats
    __shared__ __align__(16) _Float16 sWhj[TJ][SWL];   // j-rows x feats
    __shared__ __align__(16) _Float16 sWhT[NF][SCL];   // feats x j-rows
    __shared__ __align__(16) _Float16 sC[TI][SCL];     // combined att tile

    const int tid   = threadIdx.x;
    const int w     = tid >> 5;     // wave32
    const int lane  = tid & 31;
    const int nl    = lane & 15;
    const int hs    = lane >> 4;
    const int ibase = blockIdx.x * TI;

    // layer scalars: c = softmax(cf), s = softplus(sp)
    const float dc0 = dcP[0], dc1 = dcP[1];
    float f0 = cfP[0], f1 = cfP[1], f2 = cfP[2];
    float mx = fmaxf(f0, fmaxf(f1, f2));
    float e0 = __expf(f0 - mx), e1 = __expf(f1 - mx), e2 = __expf(f2 - mx);
    float inv = fast_rcp(e0 + e1 + e2);
    const float c0 = e0 * inv, c1 = e1 * inv, c2 = e2 * inv;
    const float sgl = fmaxf(spP[0], 0.f) + log1pf(__expf(-fabsf(spP[0])));

    // E tile owned by this wave: (TI/16=2) x (TJ/16=4) tiles -> 8 waves
    const int ti = w >> 2;
    const int tj = w & 3;
    // aggregation output tile
    const bool hasAgg = (w < 2 * FT);
    const int  ti2    = hasAgg ? (w / FT) : 0;
    const int  fn     = hasAgg ? (w % FT) : 0;

    // stage Wh_i tile once
    for (int cidx = tid; cidx < TI * KP / 8; cidx += 256) {
        int row = cidx / (KP / 8);
        int off = (cidx % (KP / 8)) * 8;
        *(uint4*)&sWhi[row][off] =
            *(const uint4*)&whf16[(size_t)(ibase + row) * WHLD + off];
    }

    // per-lane row constants for the elementwise stage
    float rinvI[8], adI[8];
#pragma unroll
    for (int v = 0; v < 8; ++v) {
        int ig   = ibase + ti * 16 + hs * 8 + v;
        rinvI[v] = invroot[ig];
        adI[v]   = adjdiag[ig];
    }

    v8f acc = {};
    __syncthreads();

    for (int jt = 0; jt < NJ; ++jt) {
        const int jbase = jt * TJ;

        // stage Wh_j tile (normal layout: E's B operand)
        for (int cidx = tid; cidx < TJ * KP / 8; cidx += 256) {
            int row = cidx / (KP / 8);
            int off = (cidx % (KP / 8)) * 8;
            *(uint4*)&sWhj[row][off] =
                *(const uint4*)&whf16[(size_t)(jbase + row) * WHLD + off];
        }
        // stage Wh_j^T tile (transposed layout: aggregation's B operand)
        for (int cidx = tid; cidx < NF * TJ / 8; cidx += 256) {
            int fr  = cidx / (TJ / 8);
            int off = (cidx % (TJ / 8)) * 8;
            *(uint4*)&sWhT[fr][off] =
                *(const uint4*)&whT16[(size_t)fr * NROWS + jbase + off];
        }
        if (jt + 1 < NJ && tid < TJ)
            __builtin_prefetch(&whf16[(size_t)(jbase + TJ + tid) * WHLD], 0, 0);
        __syncthreads();

        // ---- E = Wh_i @ Wh_j^T (this wave's 16x16 tile) ----
        v8f ec = {};
#pragma unroll
        for (int kc = 0; kc < KCH; ++kc) {
            v16h a = load_frag_a(&sWhi[ti * 16 + nl][kc * 32], hs);
            v16h b = load_frag_b(&sWhj[tj * 16 + nl][kc * 32], hs);
            ec = wmma_f16f32(a, b, ec);
        }

        // ---- elementwise: cosine / degree gate / split+combine ----
        const int   jg   = jbase + tj * 16 + nl;
        const float rqJi = invroot[jg];
#pragma unroll
        for (int v = 0; v < 8; ++v) {
            int   m    = hs * 8 + v;
            int   ig   = ibase + ti * 16 + m;
            float aij  = adj[(size_t)ig * NROWS + jg];
            float e    = ec[v] * (rinvI[v] * rqJi);
            float deg  = fmaf(adI[v], fast_rcp(fmaxf(aij, EPSF)), -1.f);
            float sc   = softplus_fast(fmaf(dc0, deg, dc1));
            float att  = (ig == jg) ? 0.f : e * aij * sc;
            float comb = fmaf(c0, fmaxf(att, 0.f), c1 * fminf(att, 0.f));
            sC[ti * 16 + m][tj * 16 + nl] = (_Float16)comb;
        }
        __syncthreads();

        // ---- acc += C_tile @ Wh_j ----
        if (hasAgg) {
#pragma unroll
            for (int kc = 0; kc < TJ / 32; ++kc) {
                v16h a = load_frag_a(&sC[ti2 * 16 + nl][kc * 32], hs);
                v16h b = load_frag_b(&sWhT[fn * 16 + nl][kc * 32], hs);
                acc = wmma_f16f32(a, b, acc);
            }
        }
        __syncthreads();
    }

    // ---- epilogue ----
    if (hasAgg) {
#pragma unroll
        for (int v = 0; v < 8; ++v) {
            int   ig  = ibase + ti2 * 16 + hs * 8 + v;
            int   ft  = fn * 16 + nl;
            float whv = whf32[(size_t)ig * NF + ft];
            float o   = sgl * (acc[v] + c2 * whv);
            if constexpr (MODE == 0) {
                size_t idx = (size_t)ig * 64 + ft;
                prev[idx]  = coeff * elu_f(o) + prev[idx];
            } else {
                outraw[(size_t)ig * NF + ft] = o;
            }
        }
    }
}

// ---------------------------------------------------------------------------
extern "C" void kernel_launch(void* const* d_in, const int* in_sizes, int n_in,
                              void* d_out, int out_size, void* d_ws,
                              size_t ws_size, hipStream_t stream) {
    (void)in_sizes; (void)n_in; (void)out_size; (void)ws_size;
    const float* x   = (const float*)d_in[0];
    const float* adj = (const float*)d_in[1];
    const float* Wf  = (const float*)d_in[2];
    const float* bf  = (const float*)d_in[3];
    const float* W0  = (const float*)d_in[4];
    const float* b0  = (const float*)d_in[5];
    const float* dc0 = (const float*)d_in[6];
    const float* cf0 = (const float*)d_in[7];
    const float* s0  = (const float*)d_in[8];
    const float* W1  = (const float*)d_in[9];
    const float* b1  = (const float*)d_in[10];
    const float* dc1 = (const float*)d_in[11];
    const float* cf1 = (const float*)d_in[12];
    const float* s1  = (const float*)d_in[13];
    const float* W2  = (const float*)d_in[14];
    const float* b2  = (const float*)d_in[15];
    const float* dc2 = (const float*)d_in[16];
    const float* cf2 = (const float*)d_in[17];
    const float* s2  = (const float*)d_in[18];

    char*      ws      = (char*)d_ws;
    float*     prev    = (float*)(ws);                         // 1 MB
    float*     whf32   = (float*)(ws + (1u << 20));            // 1 MB
    _Float16*  whf16   = (_Float16*)(ws + (2u << 20));         // 512 KB
    _Float16*  whT16   = (_Float16*)(ws + (2u << 20) + (512u << 10)); // 512 KB
    float*     invroot = (float*)(ws + (3u << 20));            // 16 KB
    float*     adjdiag = invroot + NROWS;                      // 16 KB
    float*     out     = (float*)d_out;

    const float coeff1 = logf(0.9f / 27.0f + 1.0f);  // log(DECAY/3^3 + 1)

    diag_kernel<<<16, 256, 0, stream>>>(adj, adjdiag);
    // layer_prev = elu(x @ Wf^T + bf)
    gemm_bias_kernel<512, 64, 1><<<64, 256, 0, stream>>>(x, Wf, bf, prev);

    // ---- layer 0 ----
    gemm_bias_kernel<512, 64, 0><<<64, 256, 0, stream>>>(x, W0, b0, whf32);
    postwh_kernel<64, 64><<<16, 256, 0, stream>>>(whf32, whf16, whT16, invroot);
    ggcn_layer_kernel<64, 0><<<128, 256, 0, stream>>>(
        adj, adjdiag, whf32, whf16, whT16, invroot, dc0, cf0, s0, prev, nullptr,
        1.0f);

    // ---- layer 1 ----
    gemm_bias_kernel<64, 64, 0><<<64, 256, 0, stream>>>(prev, W1, b1, whf32);
    postwh_kernel<64, 64><<<16, 256, 0, stream>>>(whf32, whf16, whT16, invroot);
    ggcn_layer_kernel<64, 0><<<128, 256, 0, stream>>>(
        adj, adjdiag, whf32, whf16, whT16, invroot, dc1, cf1, s1, prev, nullptr,
        coeff1);

    // ---- layer 2 (NCLS=16) + log_softmax ----
    gemm_bias_kernel<64, 16, 0><<<64, 256, 0, stream>>>(prev, W2, b2, whf32);
    postwh_kernel<16, 32><<<16, 256, 0, stream>>>(whf32, whf16, whT16, invroot);
    ggcn_layer_kernel<16, 1><<<128, 256, 0, stream>>>(
        adj, adjdiag, whf32, whf16, whT16, invroot, dc2, cf2, s2, prev, out,
        0.0f);
    logsoftmax_kernel<<<16, 256, 0, stream>>>(out);
}
